// LatentActionVQVAE_EMA_43731357008525
// MI455X (gfx1250) — matrix operand order repaired and checked by reading, not compiled, not run on hardware
//
#include <hip/hip_runtime.h>
#include <hip/hip_bf16.h>
#include <math.h>

// ---------------------------------------------------------------------------
// MI455X (gfx1250) implementation of LatentActionVQVAE forward.
// GEMM-shaped work (convs via implicit GEMM, attention/MLP matmuls) runs on
// V_WMMA_F32_16X16X32_F16 (f16 A/B staged in LDS, f32 accumulate).
// Staging uses branchless batched global loads (MLP-friendly, no per-element
// waitcnt) and 16B LDS vector ops for fragment reads (ds_load_b128).
// dec_cond's full-res conv+resize is fused down to the 35 sampled outputs.
// ---------------------------------------------------------------------------

typedef _Float16 h16;
typedef __attribute__((ext_vector_type(16))) _Float16 v16h;
typedef __attribute__((ext_vector_type(8)))  _Float16 v8h;
typedef __attribute__((ext_vector_type(4)))  _Float16 v4h;
typedef __attribute__((ext_vector_type(8)))  float    v8f;

union AFrag { v16h v; v8h h[2]; h16 e[16]; };
union CFrag { v8f  v; float e[8]; };

__device__ __forceinline__ float warp_sum32(float v) {
#pragma unroll
  for (int off = 16; off; off >>= 1) v += __shfl_xor(v, off, 32);
  return v;
}
__device__ __forceinline__ float warp_max32(float v) {
#pragma unroll
  for (int off = 16; off; off >>= 1) v = fmaxf(v, __shfl_xor(v, off, 32));
  return v;
}

// ---------------------------------------------------------------------------
// Generic GEMM: C[M,N] = act(A[M,K] * op(B) + bias[N]) + res[M,N]
// op(B): transB ? B is [N,K] row-major (X @ W^T) : B is [K,N].
// act: 0=none, 1=relu, 2=gelu(exact). 8 waves/block -> 32(M) x 64(N) tile.
// LDS: As row-major [32][32]; Bsf fragment-major [n][k] so each wave fragment
// is two contiguous 16B runs -> ds_load_b128.
// ---------------------------------------------------------------------------
__global__ __launch_bounds__(256) void lav_gemm_wmma(
    const float* __restrict__ A, const float* __restrict__ Bm,
    const float* __restrict__ bias, const float* __restrict__ res,
    float* __restrict__ C, int M, int N, int K, int transB, int act)
{
  __shared__ h16 As[32][32];     // [m][k]
  __shared__ h16 Bsf[64][32];    // [n][k]
  const int mb = blockIdx.y * 32, nb = blockIdx.x * 64;
  const int tid = threadIdx.x, lane = tid & 31, wave = tid >> 5;
  const int mw = (wave >> 2) << 4;   // 0 / 16
  const int nw = (wave & 3) << 4;    // 0 / 16 / 32 / 48

  // staging assignments (invariant over K loop)
  const int ar = tid >> 3;               // A row 0..31
  const int ac = (tid & 7) * 4;          // A col base (4 consecutive k)
  const int bn = tid >> 2;               // B n 0..63
  const int bk = (tid & 3) * 8;          // B k base (8 consecutive k)
  const int gmA = mb + ar;
  const int gnB = nb + bn;
  const bool okM = gmA < M;
  const bool okN = gnB < N;

  CFrag acc;
#pragma unroll
  for (int i = 0; i < 8; i++) acc.e[i] = 0.0f;

  for (int k0 = 0; k0 < K; k0 += 32) {
    // ---- stage A (batched branchless loads, one b64 LDS store) ----
    float av[4];
    bool aok[4];
#pragma unroll
    for (int i = 0; i < 4; i++) {
      int gk = k0 + ac + i;
      aok[i] = okM && (gk < K);
      size_t idx = aok[i] ? ((size_t)gmA * K + gk) : 0;
      av[i] = A[idx];
    }
    v4h pa;
#pragma unroll
    for (int i = 0; i < 4; i++) pa[i] = aok[i] ? (h16)av[i] : (h16)0.0f;
    *reinterpret_cast<v4h*>(&As[ar][ac]) = pa;

    // ---- stage B (batched branchless loads, one b128 LDS store) ----
    float bv[8];
    bool bok[8];
#pragma unroll
    for (int i = 0; i < 8; i++) {
      int gk = k0 + bk + i;
      bok[i] = okN && (gk < K);
      size_t idx = 0;
      if (bok[i]) idx = transB ? ((size_t)gnB * K + gk) : ((size_t)gk * N + gnB);
      bv[i] = Bm[idx];
    }
    v8h pb;
#pragma unroll
    for (int i = 0; i < 8; i++) pb[i] = bok[i] ? (h16)bv[i] : (h16)0.0f;
    *reinterpret_cast<v8h*>(&Bsf[bn][bk]) = pb;

    if (k0 + 32 < K && okM) __builtin_prefetch(&A[(size_t)gmA * K + k0 + 32], 0, 1);
    __syncthreads();

    // ---- fragments via 16B LDS vector loads ----
    AFrag a, b;
    {
      int m = lane & 15, kh = lane >> 4;
      a.h[0] = *reinterpret_cast<const v8h*>(&As[mw + m][kh * 8]);
      a.h[1] = *reinterpret_cast<const v8h*>(&As[mw + m][16 + kh * 8]);
      int n = lane & 15, ko = (lane >> 4) << 3;
      b.h[0] = *reinterpret_cast<const v8h*>(&Bsf[nw + n][ko]);
      b.h[1] = *reinterpret_cast<const v8h*>(&Bsf[nw + n][16 + ko]);
    }
    acc.v = __builtin_amdgcn_wmma_f32_16x16x32_f16(
        false, a.v, false, b.v, (short)0, acc.v, false, false);
    __syncthreads();
  }

  const int col   = nb + nw + (lane & 15);
  const int rbase = mb + mw + ((lane >> 4) << 3);
#pragma unroll
  for (int v = 0; v < 8; v++) {
    int row = rbase + v;
    if (row < M && col < N) {
      float r = acc.e[v];
      if (bias) r += bias[col];
      if (act == 1) r = fmaxf(r, 0.0f);
      else if (act == 2) r = 0.5f * r * (1.0f + erff(r * 0.70710678f));
      if (res) r += res[(size_t)row * N + col];
      C[(size_t)row * N + col] = r;
    }
  }
}

// ---------------------------------------------------------------------------
// Implicit-GEMM conv (NCHW). M=Cout, N=B*OH*OW, K=Cin*KH*KW.
// dil=2 + pre-flipped weights => ConvTranspose2d(k4,s2,p1).
// Output-pixel decomposition hoisted out of the K loop; gather is branchless.
// Epilogue: +bias, eval-BatchNorm, ReLU, direct NCHW scatter.
// ---------------------------------------------------------------------------
__global__ __launch_bounds__(256) void lav_conv2d_wmma(
    const float* __restrict__ x, const float* __restrict__ w,
    const float* __restrict__ bias, const float* __restrict__ bng,
    const float* __restrict__ bnb, float* __restrict__ y,
    int Bn, int Cin, int H, int W, int Cout,
    int KH, int KW, int sH, int sW, int pH, int pW, int dil,
    int OH, int OW, int act)
{
  __shared__ h16 As[32][32];     // [cout][k]
  __shared__ h16 Bsf[64][32];    // [n][k]
  const int KHW = KH * KW;
  const int K = Cin * KHW;
  const int OHOW = OH * OW;
  const long N = (long)Bn * OHOW;
  const int mb = blockIdx.y * 32;
  const long nb = (long)blockIdx.x * 64;
  const int tid = threadIdx.x, lane = tid & 31, wave = tid >> 5;
  const int mw = (wave >> 2) << 4;
  const int nw = (wave & 3) << 4;

  const int ar = tid >> 3;
  const int ac = (tid & 7) * 4;
  const int bn = tid >> 2;
  const int bk = (tid & 3) * 8;
  const int gmA = mb + ar;
  const bool okM = gmA < Cout;

  // hoist output-pixel decomposition (invariant over K loop)
  const long gnB = nb + bn;
  const bool okN = gnB < N;
  int pb_ = 0, poh = 0, pow_ = 0;
  if (okN) {
    pb_  = (int)(gnB / OHOW);
    int sp = (int)(gnB % OHOW);
    poh  = sp / OW;
    pow_ = sp % OW;
  }
  const int ihd0 = poh * sH - pH;
  const int iwd0 = pow_ * sW - pW;
  const float* xb = x + (size_t)pb_ * Cin * H * W;

  CFrag acc;
#pragma unroll
  for (int i = 0; i < 8; i++) acc.e[i] = 0.0f;

  for (int k0 = 0; k0 < K; k0 += 32) {
    // ---- stage weights ----
    float av[4];
    bool aok[4];
#pragma unroll
    for (int i = 0; i < 4; i++) {
      int gk = k0 + ac + i;
      aok[i] = okM && (gk < K);
      size_t idx = aok[i] ? ((size_t)gmA * K + gk) : 0;
      av[i] = w[idx];
    }
    v4h pa;
#pragma unroll
    for (int i = 0; i < 4; i++) pa[i] = aok[i] ? (h16)av[i] : (h16)0.0f;
    *reinterpret_cast<v4h*>(&As[ar][ac]) = pa;

    // ---- stage implicit-im2col tile (batched branchless gather) ----
    float bv[8];
    bool bok[8];
#pragma unroll
    for (int i = 0; i < 8; i++) {
      int gk = k0 + bk + i;
      bool ok = okN && (gk < K);
      int ci = gk / KHW, rem = gk % KHW;
      int ky = rem / KW, kx = rem % KW;
      int ihd = ihd0 + ky;
      int iwd = iwd0 + kx;
      ok = ok && ihd >= 0 && iwd >= 0 && (ihd % dil) == 0 && (iwd % dil) == 0;
      int ih = ihd / dil, iw = iwd / dil;
      ok = ok && ih < H && iw < W;
      size_t idx = ok ? (((size_t)ci * H + ih) * W + iw) : 0;
      bv[i] = xb[idx];
      bok[i] = ok;
    }
    v8h pb;
#pragma unroll
    for (int i = 0; i < 8; i++) pb[i] = bok[i] ? (h16)bv[i] : (h16)0.0f;
    *reinterpret_cast<v8h*>(&Bsf[bn][bk]) = pb;

    if (k0 + 32 < K && okM) __builtin_prefetch(&w[(size_t)gmA * K + k0 + 32], 0, 1);
    __syncthreads();

    AFrag a, b;
    {
      int m = lane & 15, kh = lane >> 4;
      a.h[0] = *reinterpret_cast<const v8h*>(&As[mw + m][kh * 8]);
      a.h[1] = *reinterpret_cast<const v8h*>(&As[mw + m][16 + kh * 8]);
      int n = lane & 15, ko = (lane >> 4) << 3;
      b.h[0] = *reinterpret_cast<const v8h*>(&Bsf[nw + n][ko]);
      b.h[1] = *reinterpret_cast<const v8h*>(&Bsf[nw + n][16 + ko]);
    }
    acc.v = __builtin_amdgcn_wmma_f32_16x16x32_f16(
        false, a.v, false, b.v, (short)0, acc.v, false, false);
    __syncthreads();
  }

  const long col  = nb + nw + (lane & 15);
  const int rbase = mb + mw + ((lane >> 4) << 3);
  if (col < N) {
    int b  = (int)(col / OHOW);
    int sp = (int)(col % OHOW);
#pragma unroll
    for (int v = 0; v < 8; v++) {
      int row = rbase + v;
      if (row < Cout) {
        float r = acc.e[v];
        if (bias) r += bias[row];
        if (bng) r = r * (bng[row] * rsqrtf(1.0f + 1e-5f)) + bnb[row];
        if (act == 1) r = fmaxf(r, 0.0f);
        y[((size_t)b * Cout + row) * OHOW + sp] = r;
      }
    }
  }
}

// -------------------------- small helper kernels ---------------------------

// [B,C,H,W] -> [B,C,W,H]
__global__ void lav_transpose_hw(const float* __restrict__ x, float* __restrict__ y,
                                 int Bn, int C, int H, int W)
{
  long i = (long)blockIdx.x * blockDim.x + threadIdx.x;
  long total = (long)Bn * C * H * W;
  if (i >= total) return;
  int w_ = (int)(i % W); long t = i / W;
  int h  = (int)(t % H); t /= H;
  int c  = (int)(t % C); int b = (int)(t / C);
  y[(((size_t)b * C + c) * W + w_) * H + h] = x[i];
}

// LayerNorm over last dim 128; one wave per row.
__global__ void lav_layernorm128(const float* __restrict__ x,
                                 const float* __restrict__ g,
                                 const float* __restrict__ b,
                                 float* __restrict__ y, int R)
{
  int r = blockIdx.x;
  if (r >= R) return;
  int lane = threadIdx.x;
  float v[4];
  float s = 0.0f;
#pragma unroll
  for (int j = 0; j < 4; j++) { v[j] = x[(size_t)r * 128 + lane + 32 * j]; s += v[j]; }
  s = warp_sum32(s);
  float mean = s * (1.0f / 128.0f);
  float q = 0.0f;
#pragma unroll
  for (int j = 0; j < 4; j++) { float d = v[j] - mean; q += d * d; }
  q = warp_sum32(q);
  float inv = rsqrtf(q * (1.0f / 128.0f) + 1e-5f);
#pragma unroll
  for (int j = 0; j < 4; j++) {
    int c = lane + 32 * j;
    y[(size_t)r * 128 + c] = (v[j] - mean) * inv * g[c] + b[c];
  }
}

// Build tq [B,36,128] = [cls|feat_p tokens] + pos, tkv [B,35,128] = feat_t tokens
__global__ void lav_build_tokens(const float* __restrict__ featp,
                                 const float* __restrict__ featt,
                                 const float* __restrict__ cls,
                                 const float* __restrict__ pos,
                                 float* __restrict__ tq, float* __restrict__ tkv,
                                 int Bn)
{
  long i = (long)blockIdx.x * blockDim.x + threadIdx.x;
  long total = (long)Bn * 36 * 128;
  if (i >= total) return;
  int c = (int)(i & 127);
  int t = (int)((i >> 7) % 36);
  int b = (int)(i / (36 * 128));
  float pv = pos[t * 128 + c];
  float val = (t == 0) ? (cls[c] + pv)
                       : (featp[((size_t)b * 128 + c) * 35 + (t - 1)] + pv);
  tq[i] = val;
  if (t < 35)
    tkv[((size_t)b * 35 + t) * 128 + c] = featt[((size_t)b * 128 + c) * 35 + t];
}

// Per-(b,h,q) attention: scores -> softmax -> weighted V. One wave per query.
__global__ void lav_attn(const float* __restrict__ Q, const float* __restrict__ K,
                         const float* __restrict__ V, float* __restrict__ O)
{
  __shared__ float sh[35];
  int t = blockIdx.x;
  int q = t % 36; int h = (t / 36) & 3; int b = t / (36 * 4);
  int lane = threadIdx.x;
  const float scale = 0.17677669529663687f;  // 1/sqrt(32)
  float qd = Q[((size_t)b * 36 + q) * 128 + h * 32 + lane];
  for (int k = 0; k < 35; k++) {
    float p = qd * K[((size_t)b * 35 + k) * 128 + h * 32 + lane];
    p = warp_sum32(p);
    if (lane == 0) sh[k] = p * scale;
  }
  __syncthreads();
  float m = -3.4e38f;
  for (int k = lane; k < 35; k += 32) m = fmaxf(m, sh[k]);
  m = warp_max32(m);
  float ssum = 0.0f;
  for (int k = lane; k < 35; k += 32) {
    float e = __expf(sh[k] - m);
    sh[k] = e; ssum += e;
  }
  ssum = warp_sum32(ssum);
  __syncthreads();
  float o = 0.0f;
  for (int k = 0; k < 35; k++)
    o += sh[k] * V[((size_t)b * 35 + k) * 128 + h * 32 + lane];
  O[((size_t)b * 36 + q) * 128 + h * 32 + lane] = o / ssum;
}

// VQ: nearest codebook entry per token; qz in [B,128,35] layout; idx+loss out.
__global__ void lav_vq(const float* __restrict__ tq, const float* __restrict__ cb,
                       float* __restrict__ qz, float* __restrict__ idx_out,
                       float* __restrict__ lossacc)
{
  int row = blockIdx.x;          // b*35 + t
  int lane = threadIdx.x;
  int b = row / 35, t = row % 35;
  const float* z = tq + ((size_t)b * 36 + 1 + t) * 128;
  float zv[4];
#pragma unroll
  for (int j = 0; j < 4; j++) zv[j] = z[lane + 32 * j];
  float bestd = 3.4e38f; int bestk = 0;
  for (int k = 0; k < 32; k++) {
    const float* ck = cb + (size_t)k * 128;
    float d = 0.0f;
#pragma unroll
    for (int j = 0; j < 4; j++) { float df = zv[j] - ck[lane + 32 * j]; d += df * df; }
    d = warp_sum32(d);
    if (d < bestd) { bestd = d; bestk = k; }
  }
  const float* cbest = cb + (size_t)bestk * 128;
#pragma unroll
  for (int j = 0; j < 4; j++)
    qz[((size_t)b * 128 + lane + 32 * j) * 35 + t] = cbest[lane + 32 * j];
  if (lane == 0) {
    idx_out[row] = (float)bestk;
    atomicAdd(lossacc, bestd);
  }
}

__global__ void lav_zero1(float* p) { if (threadIdx.x == 0 && blockIdx.x == 0) p[0] = 0.0f; }

__global__ void lav_vq_finalize(const float* __restrict__ acc, float* __restrict__ out)
{
  if (threadIdx.x == 0 && blockIdx.x == 0) {
    float s = acc[0] * (1.0f / 71680.0f);   // mean over B*128*5*7
    out[0] = 0.25f * s;                      // commit loss
    out[1] = s;                              // codebook loss
  }
}

// dec_cond conv1 at only the 20x28 sampled positions -> c1s [16,64,20,28]
__global__ void lav_dec_cond1(const float* __restrict__ x, const float* __restrict__ w,
                              const float* __restrict__ bias, float* __restrict__ y)
{
  long i = (long)blockIdx.x * blockDim.x + threadIdx.x;
  if (i >= 16L * 64 * 20 * 28) return;
  int cj = (int)(i % 28); long t = i / 28;
  int ri = (int)(t % 20); t /= 20;
  int co = (int)(t % 64); int b = (int)(t / 64);
  int yy = 32 * (ri >> 2) + 14 + (ri & 3);
  int xx = 30 * (cj >> 2) + 13 + (cj & 3);
  float acc = bias[co];
  for (int ci = 0; ci < 3; ci++)
    for (int ky = 0; ky < 3; ky++)
      for (int kx = 0; kx < 3; kx++)
        acc += w[((co * 3 + ci) * 3 + ky) * 3 + kx] *
               x[(((size_t)b * 3 + ci) * 160 + (yy - 1 + ky)) * 210 + (xx - 1 + kx)];
  y[i] = fmaxf(acc, 0.0f);
}

// dec_cond conv2 fused with 160x210 -> 5x7 bilinear resize -> cres [16,128,35]
__global__ void lav_dec_cond2(const float* __restrict__ c1s, const float* __restrict__ w,
                              const float* __restrict__ bias, float* __restrict__ y)
{
  long i = (long)blockIdx.x * blockDim.x + threadIdx.x;
  if (i >= 16L * 128 * 35) return;
  int s  = (int)(i % 35);
  int co = (int)((i / 35) % 128);
  int b  = (int)(i / (35 * 128));
  int oy = s / 7, ox = s % 7;
  float out = 0.0f;
  for (int dr = 0; dr < 2; dr++)
    for (int dc = 0; dc < 2; dc++) {
      float acc = bias[co];
      for (int ci = 0; ci < 64; ci++) {
        const float* p  = c1s + (((size_t)b * 64 + ci) * 20 + oy * 4 + dr) * 28 + ox * 4 + dc;
        const float* wp = w + ((size_t)co * 64 + ci) * 9;
        for (int ky = 0; ky < 3; ky++)
          for (int kx = 0; kx < 3; kx++)
            acc += wp[ky * 3 + kx] * p[ky * 28 + kx];
      }
      out += fmaxf(acc, 0.0f);
    }
  y[i] = 0.25f * out;
}

// Concat qz [16,128,35] and cres [16,128,35] -> xcat [16,256,35]
__global__ void lav_concat(const float* __restrict__ qz, const float* __restrict__ cres,
                           float* __restrict__ xcat)
{
  long i = (long)blockIdx.x * blockDim.x + threadIdx.x;
  if (i >= 16L * 256 * 35) return;
  int s = (int)(i % 35);
  int c = (int)((i / 35) % 256);
  int b = (int)(i / (35 * 256));
  xcat[i] = (c < 128) ? qz[((size_t)b * 128 + c) * 35 + s]
                      : cres[((size_t)b * 128 + (c - 128)) * 35 + s];
}

// ConvTranspose2d weight prep: (Cin,Cout,4,4) -> flipped OIHW [Cout, Cin*16]
__global__ void lav_convt_prep(const float* __restrict__ w, float* __restrict__ o,
                               int Cin, int Cout)
{
  long i = (long)blockIdx.x * blockDim.x + threadIdx.x;
  long total = (long)Cin * Cout * 16;
  if (i >= total) return;
  int t  = (int)(i & 15);
  int ci = (int)((i >> 4) % Cin);
  int co = (int)(i / (16L * Cin));
  int ky = t >> 2, kx = t & 3;
  o[i] = w[(((size_t)ci * Cout + co) * 4 + (3 - ky)) * 4 + (3 - kx)];
}

// Bilinear resize (antialias=False), NCHW.
__global__ void lav_resize(const float* __restrict__ x, float* __restrict__ y,
                           int Bn, int C, int Hi, int Wi, int Ho, int Wo)
{
  long i = (long)blockIdx.x * blockDim.x + threadIdx.x;
  long total = (long)Bn * C * Ho * Wo;
  if (i >= total) return;
  int ow = (int)(i % Wo); long t = i / Wo;
  int oh = (int)(t % Ho); t /= Ho;
  int c  = (int)(t % C);  int b = (int)(t / C);
  float fy = (oh + 0.5f) * (float)Hi / (float)Ho - 0.5f;
  float fx = (ow + 0.5f) * (float)Wi / (float)Wo - 0.5f;
  int y0 = (int)floorf(fy), x0 = (int)floorf(fx);
  float wy = fy - y0, wx = fx - x0;
  int y1 = y0 + 1, x1 = x0 + 1;
  y0 = max(0, min(Hi - 1, y0)); y1 = max(0, min(Hi - 1, y1));
  x0 = max(0, min(Wi - 1, x0)); x1 = max(0, min(Wi - 1, x1));
  const float* p = x + ((size_t)b * C + c) * Hi * Wi;
  float v = (1.0f - wy) * ((1.0f - wx) * p[y0 * Wi + x0] + wx * p[y0 * Wi + x1]) +
            wy * ((1.0f - wx) * p[y1 * Wi + x0] + wx * p[y1 * Wi + x1]);
  y[i] = v;
}

// ---------------------------------------------------------------------------
extern "C" void kernel_launch(void* const* d_in, const int* in_sizes, int n_in,
                              void* d_out, int out_size, void* d_ws, size_t ws_size,
                              hipStream_t stream)
{
  (void)in_sizes; (void)n_in; (void)out_size; (void)ws_size;
  auto F = [&](int i) -> const float* { return (const float*)d_in[i]; };

  // ---- input leaf mapping (jax pytree: dicts sorted by key) ----
  const float* frame_t = F(0);
  const float* frame_p = F(1);
  auto blkP = [&](int i, int j) -> const float* { return F(2 + i * 14 + j); };
  // per block j: 0=attn.bi 1=attn.bo 2=attn.wi 3=attn.wo 4=ln1kv_g 5=ln1kv_b
  //              6=ln1q_g 7=ln1q_b 8=ln2_g 9=ln2_b 10=w1 11=b1 12=w2 13=b2
  const float* cls = F(58); const float* codebook = F(59);
  const float* dcw0 = F(60); const float* dcb0 = F(61);
  const float* dcw1 = F(62); const float* dcb1 = F(63);
  const float* fcw = F(64);  const float* fcb = F(65);
  const float* fiw = F(66);  const float* fib = F(67);   // dec_final
  const float* law = F(68);  const float* lab = F(69);   // dec_last
  auto upw = [&](int i) { return F(70 + 4 * i); };
  auto upb = [&](int i) { return F(71 + 4 * i); };
  auto upg = [&](int i) { return F(72 + 4 * i); };
  auto upe = [&](int i) { return F(73 + 4 * i); };
  const float* pos = F(86);
  auto stw = [&](int i) { return F(87 + 2 * i); };
  auto stb = [&](int i) { return F(88 + 2 * i); };

  // ---- workspace bump allocator ----
  size_t off = 0;
  auto alloc = [&](size_t elems) -> float* {
    float* p = (float*)((char*)d_ws + off);
    off += ((elems * 4) + 255) & ~(size_t)255;
    return p;
  };
  float* ftT   = alloc(1612800);          // [16,3,160,210]
  float* fpT   = alloc(1612800);
  float* feat_t= alloc(71680);            // [16,128,35]
  float* feat_p= alloc(71680);
  float* tq    = alloc(73728);            // [16,36,128]
  float* tkv   = alloc(71680);            // [16,35,128]
  float* lnq   = alloc(73728);
  float* lnkv  = alloc(71680);
  float* Qb    = alloc(73728);
  float* Kb    = alloc(71680);
  float* Vb    = alloc(71680);
  float* attno = alloc(73728);
  float* hb    = alloc(294912);           // [576,512]
  float* qz    = alloc(71680);            // [16,128,35]
  float* cres  = alloc(71680);
  float* c1s   = alloc(573440);           // [16,64,20,28]
  float* xcat  = alloc(143360);           // [16,256,35]
  float* d0    = alloc(286720);           // [16,512,35]
  float* wprep = alloc(4194304);          // up to 512*512*16
  float* lossacc = alloc(64);
  float* s0    = alloc(10000000);         // ping
  float* s1    = alloc(10000000);         // pong

  float* out   = (float*)d_out;
  float* idx_out  = out + 1612800;
  float* loss_out = out + 1613360;

  auto g1 = [&](long n) { return dim3((unsigned)((n + 255) / 256)); };

  auto conv = [&](const float* x, const float* w, const float* bias,
                  const float* bng, const float* bnb, float* y,
                  int Cin, int H, int W, int Cout, int KH, int KW,
                  int sH, int sW, int pH, int pW, int dil,
                  int OH, int OW, int act) {
    long N = 16L * OH * OW;
    dim3 gr((unsigned)((N + 63) / 64), (unsigned)((Cout + 31) / 32));
    lav_conv2d_wmma<<<gr, 256, 0, stream>>>(x, w, bias, bng, bnb, y, 16, Cin, H, W,
                                            Cout, KH, KW, sH, sW, pH, pW, dil, OH, OW, act);
  };
  auto gemm = [&](const float* A, const float* B, const float* bias,
                  const float* res, float* C, int M, int N, int K, int act) {
    dim3 gr((unsigned)((N + 63) / 64), (unsigned)((M + 31) / 32));
    lav_gemm_wmma<<<gr, 256, 0, stream>>>(A, B, bias, res, C, M, N, K, 1, act);
  };

  // ---- frame transposes: [16,3,210,160] -> [16,3,160,210] ----
  lav_transpose_hw<<<g1(1612800), 256, 0, stream>>>(frame_t, ftT, 16, 3, 210, 160);
  lav_transpose_hw<<<g1(1612800), 256, 0, stream>>>(frame_p, fpT, 16, 3, 210, 160);

  // ---- conv stem (shared weights) on both frames ----
  const float* stemIn[2] = { ftT, fpT };
  float* stemOut[2] = { feat_t, feat_p };
  for (int f = 0; f < 2; f++) {
    conv(stemIn[f], stw(0), stb(0), nullptr, nullptr, s0, 3, 160, 210, 64, 4, 4, 2, 2, 1, 1, 1, 80, 105, 1);
    conv(s0, stw(1), stb(1), nullptr, nullptr, s1, 64, 80, 105, 128, 4, 4, 2, 2, 1, 1, 1, 40, 52, 1);
    conv(s1, stw(2), stb(2), nullptr, nullptr, s0, 128, 40, 52, 128, 4, 4, 2, 2, 1, 1, 1, 20, 26, 1);
    conv(s0, stw(3), stb(3), nullptr, nullptr, s1, 128, 20, 26, 128, 4, 4, 2, 2, 1, 1, 1, 10, 13, 1);
    conv(s1, stw(4), stb(4), nullptr, nullptr, stemOut[f], 128, 10, 13, 128, 4, 7, 2, 2, 1, 3, 1, 5, 7, 0);
  }

  // ---- tokens ----
  lav_build_tokens<<<g1(16L * 36 * 128), 256, 0, stream>>>(feat_p, feat_t, cls, pos, tq, tkv, 16);

  // ---- transformer blocks ----
  for (int i = 0; i < 4; i++) {
    const float* wi = blkP(i, 2); const float* bi = blkP(i, 0);
    lav_layernorm128<<<576, 32, 0, stream>>>(tq,  blkP(i, 6), blkP(i, 7), lnq, 576);
    lav_layernorm128<<<560, 32, 0, stream>>>(tkv, blkP(i, 4), blkP(i, 5), lnkv, 560);
    gemm(lnq,  wi,                 bi,       nullptr, Qb, 576, 128, 128, 0);
    gemm(lnkv, wi + 128 * 128,     bi + 128, nullptr, Kb, 560, 128, 128, 0);
    gemm(lnkv, wi + 256 * 128,     bi + 256, nullptr, Vb, 560, 128, 128, 0);
    lav_attn<<<16 * 4 * 36, 32, 0, stream>>>(Qb, Kb, Vb, attno);
    gemm(attno, blkP(i, 3), blkP(i, 1), tq, tq, 576, 128, 128, 0);   // out-proj + residual
    lav_layernorm128<<<576, 32, 0, stream>>>(tq, blkP(i, 8), blkP(i, 9), lnq, 576);
    gemm(lnq, blkP(i, 10), blkP(i, 11), nullptr, hb, 576, 512, 128, 2);  // gelu
    gemm(hb,  blkP(i, 12), blkP(i, 13), tq,      tq, 576, 128, 512, 0);  // + residual
  }

  // ---- VQ (forward value of straight-through = qz) ----
  lav_zero1<<<1, 32, 0, stream>>>(lossacc);
  lav_vq<<<560, 32, 0, stream>>>(tq, codebook, qz, idx_out, lossacc);
  lav_vq_finalize<<<1, 32, 0, stream>>>(lossacc, loss_out);

  // ---- dec_cond fused with destructive 5x7 resize ----
  lav_dec_cond1<<<g1(16L * 64 * 20 * 28), 256, 0, stream>>>(ftT, dcw0, dcb0, c1s);
  lav_dec_cond2<<<g1(16L * 128 * 35), 256, 0, stream>>>(c1s, dcw1, dcb1, cres);
  lav_concat<<<g1(16L * 256 * 35), 256, 0, stream>>>(qz, cres, xcat);

  // ---- decoder ----
  conv(xcat, fcw, fcb, nullptr, nullptr, d0, 256, 5, 7, 512, 1, 1, 1, 1, 0, 0, 1, 5, 7, 0);

  const int hd[5] = { 512, 512, 256, 128, 64 };
  const int uh[5] = { 5, 10, 20, 40, 80 };
  const int uwd[5] = { 7, 14, 28, 56, 112 };
  float* ubuf[5] = { d0, s0, s1, s0, s1 };
  for (int i = 0; i < 4; i++) {
    long pn = (long)hd[i] * hd[i + 1] * 16;
    lav_convt_prep<<<g1(pn), 256, 0, stream>>>(upw(i), wprep, hd[i], hd[i + 1]);
    conv(ubuf[i], wprep, upb(i), upg(i), upe(i), ubuf[i + 1],
         hd[i], uh[i], uwd[i], hd[i + 1], 4, 4, 1, 1, 2, 2, 2, uh[i + 1], uwd[i + 1], 1);
  }
  // dec_last: ConvT 64->3, [16,3,160,224]
  lav_convt_prep<<<g1(64L * 3 * 16), 256, 0, stream>>>(law, wprep, 64, 3);
  conv(s1, wprep, lab, nullptr, nullptr, s0, 64, 80, 112, 3, 4, 4, 1, 1, 2, 2, 2, 160, 224, 0);
  // resize 224 -> 210 (width only)
  lav_resize<<<g1(16L * 3 * 160 * 210), 256, 0, stream>>>(s0, s1, 16, 3, 160, 224, 160, 210);
  // dec_final 3x3 conv
  conv(s1, fiw, fib, nullptr, nullptr, s0, 3, 160, 210, 3, 3, 3, 1, 1, 1, 1, 1, 160, 210, 0);
  // recon = transpose back to [16,3,210,160]
  lav_transpose_hw<<<g1(1612800), 256, 0, stream>>>(s0, out, 16, 3, 160, 210);
}
